// IEGMN_Layer_60730837565977
// MI455X (gfx1250) — matrix operand order repaired and checked by reading, not compiled, not run on hardware
//
#include <hip/hip_runtime.h>
#include <hip/hip_bf16.h>
#include <stdint.h>

typedef __bf16 bf16;
typedef __attribute__((ext_vector_type(16))) __bf16 v16bf;
typedef __attribute__((ext_vector_type(8)))  __bf16 v8bf;
typedef __attribute__((ext_vector_type(8)))  float  v8f;
typedef __attribute__((ext_vector_type(4)))  unsigned int v4u;
typedef __attribute__((ext_vector_type(8)))  int v8i;
typedef __attribute__((ext_vector_type(4)))  int v4i;

#define N_LIG 4096
#define N_REC 16384
#define E_LIG 65536
#define E_REC 262144
#define HDIM  256
#define EINP  544    // 543 padded to multiple of 32
#define NINP  576
#define CH    32768  // edge chunk size

// ---------------------------------------------------------------- WMMA helpers

static __device__ inline v8f wmma_bf(v16bf a, v16bf b, v8f c) {
  return __builtin_amdgcn_wmma_f32_16x16x32_bf16(false, a, false, b, (short)0, c,
                                                 false, false);
}

// A fragment (16x32 bf16, MxK). lane 0-15: row=lane, K kk..kk+7 & kk+16..kk+23 (kk=0)
// lane 16-31: same rows, kk=8.
static __device__ inline v16bf load_a_frag(const bf16* A, int lda, int m0, int k0,
                                           int lane) {
  int row = m0 + (lane & 15);
  int kk  = k0 + ((lane & 16) ? 8 : 0);
  const v8bf* p0 = reinterpret_cast<const v8bf*>(A + (size_t)row * lda + kk);
  const v8bf* p1 = reinterpret_cast<const v8bf*>(A + (size_t)row * lda + kk + 16);
  v8bf lo = *p0, hi = *p1;
  v16bf r;
#pragma unroll
  for (int i = 0; i < 8; ++i) { r[i] = lo[i]; r[i + 8] = hi[i]; }
  return r;
}

// B fragment (32x16 bf16, KxN). lane: col=n0+lane%16, K kk..kk+7 & kk+16..kk+23 (kk=k0+8*(lane/16)).
static __device__ inline v16bf load_b_frag(const bf16* B, int ldb, int k0, int n0,
                                           int lane) {
  int col = n0 + (lane & 15);
  int kk  = k0 + ((lane & 16) ? 8 : 0);
  v16bf r;
#pragma unroll
  for (int i = 0; i < 8; ++i) {
    r[i]     = B[(size_t)(kk + i) * ldb + col];
    r[i + 8] = B[(size_t)(kk + 16 + i) * ldb + col];
  }
  return r;
}

// ---------------------------------------------------------------- TDM helper
// DMA a 32(K) x 64(N) bf16 panel from global (row stride ldb elems) into LDS.
// D# per CDNA5 ISA ch.8: group0 = {count/flags, lds_addr, global_addr, type=2},
// group1 = {mask/data_size, dims/tiles/strides}. Groups 2/3 zero (2D tensor).
static __device__ inline void tdm_load_b_panel(const bf16* gsrc, unsigned lds_off,
                                               int ldb) {
  unsigned long long ga = (unsigned long long)(uintptr_t)gsrc;
  v4u g0;
  g0[0] = 1u;                                      // count=1, user mode
  g0[1] = lds_off;                                 // lds_addr (bytes)
  g0[2] = (unsigned)(ga & 0xffffffffu);            // global_addr[31:0]
  g0[3] = (unsigned)((ga >> 32) & 0x01ffffffu)     // global_addr[56:32]
        | (2u << 30);                              // type = 2 ("image")
  v8i g1;
  g1[0] = (int)(1u << 16);        // data_size=1 -> 2 bytes/elem
  g1[1] = (int)(64u << 16);       // tensor_dim0 = 64 (bits 63:48)
  g1[2] = (int)(32u << 16);       // tensor_dim0 hi=0; tensor_dim1 = 32 (bits 95:80)
  g1[3] = (int)(64u << 16);       // tensor_dim1 hi=0; tile_dim0 = 64 (bits 127:112)
  g1[4] = 32;                     // tile_dim1 = 32; tile_dim2 = 0
  g1[5] = ldb;                    // tensor_dim0_stride (elems), low 32
  g1[6] = 0;                      // stride hi / dim1_stride lo
  g1[7] = 0;
  v4i gz = {0, 0, 0, 0};
#if defined(__clang_major__) && (__clang_major__ >= 23)
  v8i gz8 = {0, 0, 0, 0, 0, 0, 0, 0};
  __builtin_amdgcn_tensor_load_to_lds(g0, g1, gz, gz, gz8, 0);
#else
  __builtin_amdgcn_tensor_load_to_lds(g0, g1, gz, gz, 0);
#endif
}

// ---------------------------------------------------------------- generic GEMM
// C[M,256] = act(A[M,K] @ B[K,256] + bias); optional bf16 out and/or f32 out
// with skip: v = wself*v + wskip*skip. Block = 8 waves -> 128M x 64N tile.
// B panel double-buffered in LDS via Tensor Data Mover; A direct from global.
__global__ __launch_bounds__(256)
void gemm_bf16_kernel(const bf16* __restrict__ A, int lda,
                      const bf16* __restrict__ B, int ldb,
                      const float* __restrict__ bias, int lrelu,
                      int M, int N, int K,
                      bf16* __restrict__ outb, int ldob, int coff,
                      float* __restrict__ outf, int ldof,
                      const float* __restrict__ skip, int ldsk,
                      float wself, float wskip) {
  __shared__ __attribute__((aligned(32))) bf16 bpanel[2][32][64];
  int wave = threadIdx.x >> 5, lane = threadIdx.x & 31;
  int m0 = blockIdx.y * 128 + wave * 16;
  int n0 = blockIdx.x * 64;
  bool active = (m0 < M);   // wave 0 is always active; no early return (barriers!)
  v8f acc[4];
#pragma unroll
  for (int t = 0; t < 4; ++t)
#pragma unroll
    for (int r = 0; r < 8; ++r) acc[t][r] = 0.0f;

  int nsteps = K / 32;
  if (wave == 0)
    tdm_load_b_panel(B + n0, (unsigned)(uintptr_t)&bpanel[0][0][0], ldb);

  int arow = m0 + (lane & 15);
  for (int s = 0; s < nsteps; ++s) {
    int k0 = s * 32;
    if (wave == 0) __builtin_amdgcn_s_wait_tensorcnt((short)0);
    __syncthreads();
    asm volatile("" ::: "memory");
    if (wave == 0 && s + 1 < nsteps)
      tdm_load_b_panel(B + (size_t)(k0 + 32) * ldb + n0,
                       (unsigned)(uintptr_t)&bpanel[(s + 1) & 1][0][0], ldb);
    if (active) {
      if (s + 2 < nsteps)
        __builtin_prefetch((const void*)(A + (size_t)arow * lda + k0 + 64), 0, 1);
      v16bf a = load_a_frag(A, lda, m0, k0, lane);
      const bf16* bp = &bpanel[s & 1][(lane & 16) ? 8 : 0][lane & 15];
#pragma unroll
      for (int t = 0; t < 4; ++t) {
        v16bf b;
#pragma unroll
        for (int i = 0; i < 8; ++i) {
          b[i]     = bp[i * 64 + t * 16];
          b[i + 8] = bp[(i + 16) * 64 + t * 16];
        }
        acc[t] = wmma_bf(a, b, acc[t]);
      }
    }
    __syncthreads();
  }

  if (!active) return;
  int halfl = lane & 15;
  int rbase = m0 + ((lane & 16) ? 8 : 0);
#pragma unroll
  for (int t = 0; t < 4; ++t) {
    int col = n0 + t * 16 + halfl;
    float bv = bias ? bias[col] : 0.0f;
#pragma unroll
    for (int r = 0; r < 8; ++r) {
      float v = acc[t][r] + bv;
      if (lrelu) v = (v > 0.0f) ? v : 0.01f * v;
      int row = rbase + r;
      if (skip) v = wself * v + wskip * skip[(size_t)row * ldsk + col];
      if (outb) outb[(size_t)row * ldob + coff + col] = (bf16)v;
      if (outf) outf[(size_t)row * ldof + col] = v;
    }
  }
}

// ---------------------------------------------------------------- flash attention
// one wave per 16 q rows; full 256-wide output per wave; online softmax.
__global__ __launch_bounds__(256)
void flash_attn(const bf16* __restrict__ Q, const bf16* __restrict__ KT,
                const bf16* __restrict__ V, const float* __restrict__ mask,
                int mask_ld, int maskT, int Mq, int Nk,
                bf16* __restrict__ outp, int ldo, int coff) {
  __shared__ bf16 pbuf[8][16 * 32];
  int wave = threadIdx.x >> 5, lane = threadIdx.x & 31;
  int m0 = (blockIdx.x * 8 + wave) * 16;
  if (m0 >= Mq) return;
  int halfl = lane & 15;
  int rbase = m0 + ((lane & 16) ? 8 : 0);
  v16bf qf[8];
#pragma unroll
  for (int s = 0; s < 8; ++s) qf[s] = load_a_frag(Q, 256, m0, s * 32, lane);
  v8f o[16];
#pragma unroll
  for (int t = 0; t < 16; ++t)
#pragma unroll
    for (int r = 0; r < 8; ++r) o[t][r] = 0.0f;
  float mr[8], lr[8];
#pragma unroll
  for (int r = 0; r < 8; ++r) { mr[r] = -1e30f; lr[r] = 0.0f; }
  bf16* pw = pbuf[wave];

  for (int n0 = 0; n0 < Nk; n0 += 32) {
    // prefetch next mask tile (mask is the dominant streaming read)
    if (n0 + 32 < Nk) {
      int kn = n0 + 32 + halfl;
      const float* mp = maskT ? &mask[(size_t)kn * mask_ld + rbase]
                              : &mask[(size_t)rbase * mask_ld + kn];
      __builtin_prefetch((const void*)mp, 0, 1);
    }
    v8f s0, s1;
#pragma unroll
    for (int r = 0; r < 8; ++r) { s0[r] = 0.0f; s1[r] = 0.0f; }
#pragma unroll
    for (int s = 0; s < 8; ++s)
      s0 = wmma_bf(qf[s], load_b_frag(KT, Nk, s * 32, n0, lane), s0);
#pragma unroll
    for (int s = 0; s < 8; ++s)
      s1 = wmma_bf(qf[s], load_b_frag(KT, Nk, s * 32, n0 + 16, lane), s1);

    int kc0 = n0 + halfl, kc1 = n0 + 16 + halfl;
    float p0a[8], p1a[8], sc[8];
#pragma unroll
    for (int r = 0; r < 8; ++r) {
      int qr = rbase + r;
      float mv0 = maskT ? mask[(size_t)kc0 * mask_ld + qr]
                        : mask[(size_t)qr * mask_ld + kc0];
      float mv1 = maskT ? mask[(size_t)kc1 * mask_ld + qr]
                        : mask[(size_t)qr * mask_ld + kc1];
      float a0 = mv0 * s0[r] - 1000.0f * (1.0f - mv0);
      float a1 = mv1 * s1[r] - 1000.0f * (1.0f - mv1);
      float tmax = fmaxf(a0, a1);
      for (int off = 1; off < 16; off <<= 1)
        tmax = fmaxf(tmax, __shfl_xor(tmax, off, 32));
      float mn = fmaxf(mr[r], tmax);
      float scale = __expf(mr[r] - mn);
      float p0 = __expf(a0 - mn), p1 = __expf(a1 - mn);
      float rs = p0 + p1;
      for (int off = 1; off < 16; off <<= 1) rs += __shfl_xor(rs, off, 32);
      lr[r] = lr[r] * scale + rs;
      mr[r] = mn;
      sc[r] = scale; p0a[r] = p0; p1a[r] = p1;
    }
#pragma unroll
    for (int t = 0; t < 16; ++t)
#pragma unroll
      for (int r = 0; r < 8; ++r) o[t][r] *= sc[r];
    // C-layout probs -> LDS -> A-layout fragment
#pragma unroll
    for (int r = 0; r < 8; ++r) {
      int row = r + ((lane & 16) ? 8 : 0);
      pw[row * 32 + halfl]      = (bf16)p0a[r];
      pw[row * 32 + 16 + halfl] = (bf16)p1a[r];
    }
    v16bf pa;
    {
      int row = halfl, kk = (lane & 16) ? 8 : 0;
#pragma unroll
      for (int i = 0; i < 8; ++i) {
        pa[i]     = pw[row * 32 + kk + i];
        pa[i + 8] = pw[row * 32 + kk + 16 + i];
      }
    }
#pragma unroll
    for (int t = 0; t < 16; ++t)
      o[t] = wmma_bf(pa, load_b_frag(V, 256, n0, t * 16, lane), o[t]);
  }
#pragma unroll
  for (int t = 0; t < 16; ++t) {
    int col = coff + t * 16 + halfl;
#pragma unroll
    for (int r = 0; r < 8; ++r) {
      int qr = rbase + r;
      outp[(size_t)qr * ldo + col] = (bf16)(o[t][r] / lr[r]);
    }
  }
}

// ---------------------------------------------------------------- small kernels

__global__ __launch_bounds__(256)
void cvt_pad_bf16(const float* __restrict__ src, bf16* __restrict__ dst,
                  int K, int N, int Kpad) {
  int i = blockIdx.x * 256 + threadIdx.x;
  if (i >= Kpad * N) return;
  int r = i / N;
  float v = (r < K) ? src[i] : 0.0f;
  dst[i] = (bf16)v;
}

__global__ __launch_bounds__(256)
void transpose_bf16(const bf16* __restrict__ src, bf16* __restrict__ dst,
                    int M, int N) {  // dst[N,M]
  __shared__ bf16 tile[32][33];
  int bx = blockIdx.x * 32, by = blockIdx.y * 32;
  int tx = threadIdx.x, ty = threadIdx.y;
  for (int i = ty; i < 32; i += 8) {
    int r = by + i, c = bx + tx;
    tile[i][tx] = (r < M && c < N) ? src[(size_t)r * N + c] : (bf16)0.0f;
  }
  __syncthreads();
  for (int i = ty; i < 32; i += 8) {
    int r = bx + i, c = by + tx;
    if (r < N && c < M) dst[(size_t)r * M + c] = tile[tx][i];
  }
}

// one wave per edge: build bf16 row [h_src | h_dst | ef | rbf | pad] + x_rel
__global__ __launch_bounds__(256)
void gather_edges(const float* __restrict__ coords, const float* __restrict__ h,
                  const float* __restrict__ ef, const int* __restrict__ src,
                  const int* __restrict__ dst, int e0,
                  bf16* __restrict__ Xe, float* __restrict__ xrel) {
  int wave = threadIdx.x >> 5, lane = threadIdx.x & 31;
  int le = blockIdx.x * 8 + wave;
  int e = e0 + le;
  int s = src[e], d = dst[e];
  float x0 = coords[s * 3 + 0] - coords[d * 3 + 0];
  float x1 = coords[s * 3 + 1] - coords[d * 3 + 1];
  float x2 = coords[s * 3 + 2] - coords[d * 3 + 2];
  float d2 = x0 * x0 + x1 * x1 + x2 * x2;
  if (lane == 0) {
    xrel[le * 3 + 0] = x0; xrel[le * 3 + 1] = x1; xrel[le * 3 + 2] = x2;
  }
  bf16* row = Xe + (size_t)le * EINP;
  for (int idx = lane; idx < EINP; idx += 32) {
    float v;
    if (idx < 256)      v = h[(size_t)s * 256 + idx];
    else if (idx < 512) v = h[(size_t)d * 256 + (idx - 256)];
    else if (idx < 528) v = ef[(size_t)e * 16 + (idx - 512)];
    else if (idx < 543) v = __expf(-d2 / __powf(1.5f, (float)(idx - 528)));
    else                v = 0.0f;
    row[idx] = (bf16)v;
  }
}

// per-edge scalar s = t . w2 + b2, scatter x_rel*s into xacc[dst]
__global__ __launch_bounds__(256)
void coords_scalar(const bf16* __restrict__ t, const float* __restrict__ w2,
                   const float* __restrict__ b2, const float* __restrict__ xrel,
                   const int* __restrict__ dst, int e0, float* __restrict__ xacc) {
  int wave = threadIdx.x >> 5, lane = threadIdx.x & 31;
  int le = blockIdx.x * 8 + wave;
  const bf16* tr = t + (size_t)le * 256;
  float s = 0.0f;
  for (int i = lane; i < 256; i += 32) s += (float)tr[i] * w2[i];
  for (int off = 16; off > 0; off >>= 1) s += __shfl_down(s, off, 32);
  s = __shfl(s, 0, 32) + b2[0];
  if (lane < 3) {
    int d = dst[e0 + le];
    atomicAdd(&xacc[d * 3 + lane], xrel[le * 3 + lane] * s);
  }
}

__global__ __launch_bounds__(256)
void aggregate_msg(const bf16* __restrict__ msg, const int* __restrict__ dst, int e0,
                   float* __restrict__ accum, float* __restrict__ cnt) {
  int le = blockIdx.x, tid = threadIdx.x;
  int d = dst[e0 + le];
  atomicAdd(&accum[(size_t)d * 256 + tid], (float)msg[(size_t)le * 256 + tid]);
  if (tid == 0) atomicAdd(&cnt[d], 1.0f);
}

// per node: aggr mean -> Xn[0:256], orig -> Xn[512:576], coord update -> d_out
__global__ __launch_bounds__(320)
void finalize_nodes(const float* __restrict__ accum, const float* __restrict__ cnt,
                    const float* __restrict__ orig, const float* __restrict__ xacc,
                    const float* __restrict__ coords, const float* __restrict__ origc,
                    bf16* __restrict__ Xn, float* __restrict__ xev) {
  int n = blockIdx.x, tid = threadIdx.x;
  float c = fmaxf(cnt[n], 1.0f);
  if (tid < 256)
    Xn[(size_t)n * NINP + tid] = (bf16)(accum[(size_t)n * 256 + tid] / c);
  else
    Xn[(size_t)n * NINP + 512 + (tid - 256)] = (bf16)orig[(size_t)n * 64 + (tid - 256)];
  if (tid < 3)
    xev[n * 3 + tid] = 0.25f * origc[n * 3 + tid] + 0.75f * coords[n * 3 + tid]
                     + xacc[n * 3 + tid] / c;
}

// ---------------------------------------------------------------- launch

extern "C" void kernel_launch(void* const* d_in, const int* in_sizes, int n_in,
                              void* d_out, int out_size, void* d_ws, size_t ws_size,
                              hipStream_t stream) {
  (void)in_sizes; (void)n_in; (void)out_size; (void)ws_size;
  const float* coords_lig = (const float*)d_in[0];
  const float* h_lig      = (const float*)d_in[1];
  const float* orig_lig   = (const float*)d_in[2];
  const float* origc_lig  = (const float*)d_in[3];
  const float* coords_rec = (const float*)d_in[4];
  const float* h_rec      = (const float*)d_in[5];
  const float* orig_rec   = (const float*)d_in[6];
  const float* origc_rec  = (const float*)d_in[7];
  const float* mask       = (const float*)d_in[8];
  const float* lig_ef     = (const float*)d_in[9];
  const float* rec_ef     = (const float*)d_in[10];
  const int*   lig_src    = (const int*)d_in[11];
  const int*   lig_dst    = (const int*)d_in[12];
  const int*   rec_src    = (const int*)d_in[13];
  const int*   rec_dst    = (const int*)d_in[14];
  const float* lig_e_w1   = (const float*)d_in[15];
  const float* lig_e_b1   = (const float*)d_in[16];
  const float* lig_e_w2   = (const float*)d_in[17];
  const float* lig_e_b2   = (const float*)d_in[18];
  const float* rec_e_w1   = (const float*)d_in[19];
  const float* rec_e_b1   = (const float*)d_in[20];
  const float* rec_e_w2   = (const float*)d_in[21];
  const float* rec_e_b2   = (const float*)d_in[22];
  const float* q_lig_w    = (const float*)d_in[23];
  const float* k_lig_w    = (const float*)d_in[24];
  const float* v_lig_w    = (const float*)d_in[25];
  const float* q_rec_w    = (const float*)d_in[26];
  const float* k_rec_w    = (const float*)d_in[27];
  const float* v_rec_w    = (const float*)d_in[28];
  const float* n_lig_w1   = (const float*)d_in[29];
  const float* n_lig_b1   = (const float*)d_in[30];
  const float* n_lig_w2   = (const float*)d_in[31];
  const float* n_lig_b2   = (const float*)d_in[32];
  const float* n_rec_w1   = (const float*)d_in[33];
  const float* n_rec_b1   = (const float*)d_in[34];
  const float* n_rec_w2   = (const float*)d_in[35];
  const float* n_rec_b2   = (const float*)d_in[36];
  const float* c_lig_w1   = (const float*)d_in[37];
  const float* c_lig_b1   = (const float*)d_in[38];
  const float* c_lig_w2   = (const float*)d_in[39];
  const float* c_lig_b2   = (const float*)d_in[40];
  const float* c_rec_w1   = (const float*)d_in[41];
  const float* c_rec_b1   = (const float*)d_in[42];
  const float* c_rec_w2   = (const float*)d_in[43];
  const float* c_rec_b2   = (const float*)d_in[44];

  float* outf = (float*)d_out;
  const size_t OFF_XL = 0;
  const size_t OFF_HL = (size_t)N_LIG * 3;
  const size_t OFF_XR = OFF_HL + (size_t)N_LIG * HDIM;
  const size_t OFF_HR = OFF_XR + (size_t)N_REC * 3;

  char* p = (char*)d_ws;
  auto carve = [&](size_t bytes) -> void* {
    void* r = (void*)p;
    p += (bytes + 255) & ~(size_t)255;
    return r;
  };
  auto carve_bf = [&](size_t elems) -> bf16* { return (bf16*)carve(elems * 2); };

  // fp32 accumulators (zeroed once per call with a single memset)
  char* zbase = p;
  float* accum_lig = (float*)carve((size_t)N_LIG * 256 * 4);
  float* accum_rec = (float*)carve((size_t)N_REC * 256 * 4);
  float* cnt_lig   = (float*)carve((size_t)N_LIG * 4);
  float* cnt_rec   = (float*)carve((size_t)N_REC * 4);
  float* xacc_lig  = (float*)carve((size_t)N_LIG * 3 * 4);
  float* xacc_rec  = (float*)carve((size_t)N_REC * 3 * 4);
  size_t zbytes = (size_t)(p - zbase);
  hipMemsetAsync(zbase, 0, zbytes, stream);

  float* xrel = (float*)carve((size_t)CH * 3 * 4);

  bf16* h_lig_bf = carve_bf((size_t)N_LIG * 256);
  bf16* h_rec_bf = carve_bf((size_t)N_REC * 256);
  bf16* qL  = carve_bf((size_t)N_LIG * 256);
  bf16* kL  = carve_bf((size_t)N_LIG * 256);
  bf16* vL  = carve_bf((size_t)N_LIG * 256);
  bf16* kTL = carve_bf((size_t)256 * N_LIG);
  bf16* qR  = carve_bf((size_t)N_REC * 256);
  bf16* kR  = carve_bf((size_t)N_REC * 256);
  bf16* vR  = carve_bf((size_t)N_REC * 256);
  bf16* kTR = carve_bf((size_t)256 * N_REC);
  bf16* XnL = carve_bf((size_t)N_LIG * NINP);
  bf16* XnR = carve_bf((size_t)N_REC * NINP);
  bf16* nh  = carve_bf((size_t)N_REC * 256);
  bf16* Xe    = carve_bf((size_t)CH * EINP);
  bf16* msg_h = carve_bf((size_t)CH * 256);
  bf16* msg   = carve_bf((size_t)CH * 256);
  bf16* tb    = carve_bf((size_t)CH * 256);
  bf16* WqL = carve_bf(256 * 256); bf16* WkL = carve_bf(256 * 256);
  bf16* WvL = carve_bf(256 * 256);
  bf16* WqR = carve_bf(256 * 256); bf16* WkR = carve_bf(256 * 256);
  bf16* WvR = carve_bf(256 * 256);
  bf16* We1L = carve_bf((size_t)EINP * 256); bf16* We1R = carve_bf((size_t)EINP * 256);
  bf16* We2L = carve_bf(256 * 256);          bf16* We2R = carve_bf(256 * 256);
  bf16* Wn1L = carve_bf((size_t)NINP * 256); bf16* Wn1R = carve_bf((size_t)NINP * 256);
  bf16* Wn2L = carve_bf(256 * 256);          bf16* Wn2R = carve_bf(256 * 256);
  bf16* Wc1L = carve_bf(256 * 256);          bf16* Wc1R = carve_bf(256 * 256);

  auto cvt = [&](const float* s, bf16* d, int K, int N2, int Kpad) {
    int tot = Kpad * N2;
    cvt_pad_bf16<<<(tot + 255) / 256, 256, 0, stream>>>(s, d, K, N2, Kpad);
  };
  auto gemm = [&](const bf16* A, int lda, const bf16* B, int ldb,
                  const float* bias, int lrelu, int M, int K,
                  bf16* ob, int ldob, int coff, float* of, int ldof,
                  const float* skip, int ldsk, float wself, float wskip) {
    dim3 g(4, (unsigned)((M + 127) / 128));
    gemm_bf16_kernel<<<g, 256, 0, stream>>>(A, lda, B, ldb, bias, lrelu, M, 256, K,
                                            ob, ldob, coff, of, ldof, skip, ldsk,
                                            wself, wskip);
  };

  // --- convert weights & node features to bf16 (pad 543->544)
  cvt(h_lig, h_lig_bf, N_LIG, 256, N_LIG);
  cvt(h_rec, h_rec_bf, N_REC, 256, N_REC);
  cvt(q_lig_w, WqL, 256, 256, 256); cvt(k_lig_w, WkL, 256, 256, 256);
  cvt(v_lig_w, WvL, 256, 256, 256);
  cvt(q_rec_w, WqR, 256, 256, 256); cvt(k_rec_w, WkR, 256, 256, 256);
  cvt(v_rec_w, WvR, 256, 256, 256);
  cvt(lig_e_w1, We1L, 543, 256, EINP); cvt(rec_e_w1, We1R, 543, 256, EINP);
  cvt(lig_e_w2, We2L, 256, 256, 256);  cvt(rec_e_w2, We2R, 256, 256, 256);
  cvt(n_lig_w1, Wn1L, NINP, 256, NINP); cvt(n_rec_w1, Wn1R, NINP, 256, NINP);
  cvt(n_lig_w2, Wn2L, 256, 256, 256);   cvt(n_rec_w2, Wn2R, 256, 256, 256);
  cvt(c_lig_w1, Wc1L, 256, 256, 256);   cvt(c_rec_w1, Wc1R, 256, 256, 256);

  // --- attention projections (q,k: leaky relu; v: linear)
  gemm(h_lig_bf, 256, WqL, 256, nullptr, 1, N_LIG, 256, qL, 256, 0,
       nullptr, 0, nullptr, 0, 0.f, 0.f);
  gemm(h_lig_bf, 256, WkL, 256, nullptr, 1, N_LIG, 256, kL, 256, 0,
       nullptr, 0, nullptr, 0, 0.f, 0.f);
  gemm(h_lig_bf, 256, WvL, 256, nullptr, 0, N_LIG, 256, vL, 256, 0,
       nullptr, 0, nullptr, 0, 0.f, 0.f);
  gemm(h_rec_bf, 256, WqR, 256, nullptr, 1, N_REC, 256, qR, 256, 0,
       nullptr, 0, nullptr, 0, 0.f, 0.f);
  gemm(h_rec_bf, 256, WkR, 256, nullptr, 1, N_REC, 256, kR, 256, 0,
       nullptr, 0, nullptr, 0, 0.f, 0.f);
  gemm(h_rec_bf, 256, WvR, 256, nullptr, 0, N_REC, 256, vR, 256, 0,
       nullptr, 0, nullptr, 0, 0.f, 0.f);

  transpose_bf16<<<dim3(8, N_LIG / 32), dim3(32, 8), 0, stream>>>(kL, kTL, N_LIG, 256);
  transpose_bf16<<<dim3(8, N_REC / 32), dim3(32, 8), 0, stream>>>(kR, kTR, N_REC, 256);

  // --- cross attention -> Xn columns [256, 512)
  flash_attn<<<N_LIG / 128, 256, 0, stream>>>(qL, kTR, vR, mask, N_REC, 0,
                                              N_LIG, N_REC, XnL, NINP, 256);
  flash_attn<<<N_REC / 128, 256, 0, stream>>>(qR, kTL, vL, mask, N_REC, 1,
                                              N_REC, N_LIG, XnR, NINP, 256);

  // --- ligand edges (chunks of CH)
  for (int c = 0; c < E_LIG / CH; ++c) {
    int e0 = c * CH;
    gather_edges<<<CH / 8, 256, 0, stream>>>(coords_lig, h_lig, lig_ef,
                                             lig_src, lig_dst, e0, Xe, xrel);
    gemm(Xe, EINP, We1L, 256, lig_e_b1, 1, CH, EINP, msg_h, 256, 0,
         nullptr, 0, nullptr, 0, 0.f, 0.f);
    gemm(msg_h, 256, We2L, 256, lig_e_b2, 0, CH, 256, msg, 256, 0,
         nullptr, 0, nullptr, 0, 0.f, 0.f);
    gemm(msg, 256, Wc1L, 256, c_lig_b1, 1, CH, 256, tb, 256, 0,
         nullptr, 0, nullptr, 0, 0.f, 0.f);
    coords_scalar<<<CH / 8, 256, 0, stream>>>(tb, c_lig_w2, c_lig_b2, xrel,
                                              lig_dst, e0, xacc_lig);
    aggregate_msg<<<CH, 256, 0, stream>>>(msg, lig_dst, e0, accum_lig, cnt_lig);
  }
  // --- receptor edges
  for (int c = 0; c < E_REC / CH; ++c) {
    int e0 = c * CH;
    gather_edges<<<CH / 8, 256, 0, stream>>>(coords_rec, h_rec, rec_ef,
                                             rec_src, rec_dst, e0, Xe, xrel);
    gemm(Xe, EINP, We1R, 256, rec_e_b1, 1, CH, EINP, msg_h, 256, 0,
         nullptr, 0, nullptr, 0, 0.f, 0.f);
    gemm(msg_h, 256, We2R, 256, rec_e_b2, 0, CH, 256, msg, 256, 0,
         nullptr, 0, nullptr, 0, 0.f, 0.f);
    gemm(msg, 256, Wc1R, 256, c_rec_b1, 1, CH, 256, tb, 256, 0,
         nullptr, 0, nullptr, 0, 0.f, 0.f);
    coords_scalar<<<CH / 8, 256, 0, stream>>>(tb, c_rec_w2, c_rec_b2, xrel,
                                              rec_dst, e0, xacc_rec);
    aggregate_msg<<<CH, 256, 0, stream>>>(msg, rec_dst, e0, accum_rec, cnt_rec);
  }

  // --- finalize: segment means, Xn assembly, coordinate outputs
  finalize_nodes<<<N_LIG, 320, 0, stream>>>(accum_lig, cnt_lig, orig_lig, xacc_lig,
                                            coords_lig, origc_lig, XnL, outf + OFF_XL);
  finalize_nodes<<<N_REC, 320, 0, stream>>>(accum_rec, cnt_rec, orig_rec, xacc_rec,
                                            coords_rec, origc_rec, XnR, outf + OFF_XR);

  // --- node MLPs with 0.5/0.5 skip, fp32 outputs straight into d_out
  gemm(XnL, NINP, Wn1L, 256, n_lig_b1, 1, N_LIG, NINP, nh, 256, 0,
       nullptr, 0, nullptr, 0, 0.f, 0.f);
  gemm(nh, 256, Wn2L, 256, n_lig_b2, 0, N_LIG, 256, nullptr, 0, 0,
       outf + OFF_HL, 256, h_lig, 256, 0.5f, 0.5f);
  gemm(XnR, NINP, Wn1R, 256, n_rec_b1, 1, N_REC, NINP, nh, 256, 0,
       nullptr, 0, nullptr, 0, 0.f, 0.f);
  gemm(nh, 256, Wn2R, 256, n_rec_b2, 0, N_REC, 256, nullptr, 0, 0,
       outf + OFF_HR, 256, h_rec, 256, 0.5f, 0.5f);
}
